// multiwindowSS2D_15985868275962
// MI455X (gfx1250) — matrix operand order repaired and checked by reading, not compiled, not run on hardware
//
#include <hip/hip_runtime.h>
#include <hip/hip_bf16.h>

// ---- problem constants (from reference) ----
#define D_INNER 192
#define D_STATE 16
#define DT_RANK 6
#define DD      38      // DT_RANK + 2*D_STATE
#define LW      9       // 3x3 window
#define KDIR    4
#define OUTCH_  96
#define H_      48
#define W_      48
#define HW      2304    // 48*48
#define MROWS   152     // KDIR*DD rows of the x_dbl projection

typedef __attribute__((ext_vector_type(16))) _Float16 v16h;
typedef __attribute__((ext_vector_type(8)))  float    v8f;

__device__ __forceinline__ int reflect48(int v) {
  if (v < 0)   v = -v;        // jnp 'reflect': -1 -> 1
  if (v >= 48) v = 94 - v;    // 48 -> 46
  return v;
}

// A-fragment (16x32 f16, wave32): lane&15 = M row, hi = lane>>4.
// element e<8 -> K = 8*hi + e ; e>=8 -> K = 16 + 8*hi + (e-8).
// Both runs are 8 contiguous floats -> two float4 pairs (global_load_b128).
__device__ __forceinline__ v16h load_a_frag(const float* __restrict__ rowbase,
                                            int hi, float msk) {
  const float4* q0 = reinterpret_cast<const float4*>(rowbase + 8 * hi);
  const float4* q1 = reinterpret_cast<const float4*>(rowbase + 16 + 8 * hi);
  const float4 r0 = q0[0], r1 = q0[1], r2 = q1[0], r3 = q1[1];
  v16h a;
  a[0]=(_Float16)(r0.x*msk); a[1]=(_Float16)(r0.y*msk);
  a[2]=(_Float16)(r0.z*msk); a[3]=(_Float16)(r0.w*msk);
  a[4]=(_Float16)(r1.x*msk); a[5]=(_Float16)(r1.y*msk);
  a[6]=(_Float16)(r1.z*msk); a[7]=(_Float16)(r1.w*msk);
  a[8]=(_Float16)(r2.x*msk); a[9]=(_Float16)(r2.y*msk);
  a[10]=(_Float16)(r2.z*msk); a[11]=(_Float16)(r2.w*msk);
  a[12]=(_Float16)(r3.x*msk); a[13]=(_Float16)(r3.y*msk);
  a[14]=(_Float16)(r3.z*msk); a[15]=(_Float16)(r3.w*msk);
  return a;
}

// B-fragment (32x16 f16, wave32): lane&15 = N col, hi = lane>>4.
// element e<8 -> K = 8*hi + e ; e>=8 -> K = 16 + 8*hi + (e-8) (4x b128).
__device__ __forceinline__ v16h load_b_frag(const float* __restrict__ colbase,
                                            int hi) {
  const float4* q0 = reinterpret_cast<const float4*>(colbase + 8 * hi);
  const float4* q1 = reinterpret_cast<const float4*>(colbase + 16 + 8 * hi);
  const float4 r0 = q0[0], r1 = q0[1], r2 = q1[0], r3 = q1[1];
  v16h b;
  b[0]=(_Float16)r0.x; b[1]=(_Float16)r0.y; b[2]=(_Float16)r0.z; b[3]=(_Float16)r0.w;
  b[4]=(_Float16)r1.x; b[5]=(_Float16)r1.y; b[6]=(_Float16)r1.z; b[7]=(_Float16)r1.w;
  b[8]=(_Float16)r2.x; b[9]=(_Float16)r2.y; b[10]=(_Float16)r2.z; b[11]=(_Float16)r2.w;
  b[12]=(_Float16)r3.x; b[13]=(_Float16)r3.y; b[14]=(_Float16)r3.z; b[15]=(_Float16)r3.w;
  return b;
}

// ============================================================
// Kernel 0: transpose x [192][2304] -> xt [2304][192]
// (makes all later K-dim reads contiguous / coalesced; 1.7MB, L2-resident)
// ============================================================
__global__ void __launch_bounds__(256)
transpose_x(const float* __restrict__ x, float* __restrict__ xt) {
  __shared__ float tile[16][17];
  const int bp = blockIdx.x * 16;          // pixel tile
  const int bc = blockIdx.y * 16;          // channel tile
  const int tx = threadIdx.x & 15, ty = threadIdx.x >> 4;
  tile[ty][tx] = x[(bc + ty) * HW + bp + tx];     // coalesced along pixels
  __syncthreads();
  xt[(bp + ty) * D_INNER + bc + tx] = tile[tx][ty]; // coalesced along channels
}

// ============================================================
// Kernel 1: x_dbl = x_proj_weight [152x192] @ xT^T  -> [152][2304]
// One wave32 per 16x16 tile; 6 back-to-back v_wmma_f32_16x16x32_f16.
// ============================================================
__global__ void __launch_bounds__(32)
xdbl_gemm_wmma(const float* __restrict__ xt,   // [2304][192]
               const float* __restrict__ xw,   // [152][192]
               float* __restrict__ xdbl)       // [152][2304]
{
  const int lane = threadIdx.x;
  const int m0 = blockIdx.y * 16;      // row tile (padded to 160)
  const int p0 = blockIdx.x * 16;      // pixel tile
  const int ml = lane & 15;
  const int hi = lane >> 4;

  const int arow  = m0 + ml;
  const int arowc = (arow < MROWS) ? arow : (MROWS - 1);  // clamp: load stays in-bounds
  const float msk = (arow < MROWS) ? 1.0f : 0.0f;         // zero pad rows via mask

  v8f acc = {};
#pragma unroll
  for (int kb = 0; kb < D_INNER; kb += 32) {
    const v16h a = load_a_frag(&xw[arowc * D_INNER + kb], hi, msk);
    const v16h b = load_b_frag(&xt[(p0 + ml) * D_INNER + kb], hi);
    acc = __builtin_amdgcn_wmma_f32_16x16x32_f16(false, a, false, b,
                                                 (short)0, acc, false, false);
  }
  // a lane's 8 output rows are the contiguous span m0+8*hi .. m0+8*hi+7;
  // MROWS (152) is a multiple of 8, so the span is all-valid or all-invalid:
  // one divergence point instead of 8 predicated stores.
  if (m0 + 8 * hi + 7 < MROWS) {
#pragma unroll
    for (int r = 0; r < 8; ++r)
      xdbl[(m0 + r + 8 * hi) * HW + p0 + ml] = acc[r];
  }
}

// ============================================================
// Kernel 2: fused dt-projection + 4-direction selective scan +
// direction recombination + layernorm + 3x3 conv reduce.
// One block per pixel, one thread per channel. 16-state recurrence
// in registers; shared window operands staged in LDS.
// ============================================================
__global__ void __launch_bounds__(D_INNER)
scan_ln_conv(const float* __restrict__ xt,     // [2304][192]
             const float* __restrict__ xdbl,   // [152][2304]
             const float* __restrict__ dtw,    // [4][192][6]
             const float* __restrict__ dtb,    // [4][192]
             const float* __restrict__ alogs,  // [768][16]
             const float* __restrict__ Dsv,    // [768]
             const float* __restrict__ lnw,    // [192]
             const float* __restrict__ lnb,    // [192]
             const float* __restrict__ convw,  // [192][9]
             float* __restrict__ res)          // [2304][192]
{
  __shared__ float xdw[KDIR * DD * LW];   // 4 proj x 38 ch x 9 pos
  __shared__ float red[LW * D_INNER];     // LN reduction buffer
  __shared__ float mu[LW], rs[LW];

  const int p = blockIdx.x;
  const int c = threadIdx.x;
  const int py = p / W_, px = p % W_;

  __builtin_prefetch(&xt[p * D_INNER + c], 0, 3);  // global_prefetch_b8

  // cooperative stage of x_dbl windows into LDS
  for (int idx = c; idx < KDIR * DD * LW; idx += D_INNER) {
    const int kp  = idx / (DD * LW);
    const int rem = idx - kp * (DD * LW);
    const int ch  = rem / LW;
    const int l   = rem - ch * LW;
    const int yy = reflect48(py + l / 3 - 1);
    const int xx = reflect48(px + l % 3 - 1);
    xdw[idx] = xdbl[(kp * DD + ch) * HW + yy * W_ + xx];
  }

  // this thread's u window; coalesced across channels thanks to xT layout
  float uwin[LW];
#pragma unroll
  for (int l = 0; l < LW; ++l) {
    const int yy = reflect48(py + l / 3 - 1);
    const int xx = reflect48(px + l % 3 - 1);
    uwin[l] = xt[(yy * W_ + xx) * D_INNER + c];
  }
  __syncthreads();

  float outacc[LW];
#pragma unroll
  for (int l = 0; l < LW; ++l) outacc[l] = 0.0f;

  for (int dir = 0; dir < KDIR; ++dir) {
    // direction -> source projection: {0,2,1,3} == swap low 2 bits
    const int kp = ((dir & 1) << 1) | ((dir >> 1) & 1);
    const float* xb = &xdw[kp * DD * LW];

    float wr[DT_RANK];
#pragma unroll
    for (int r = 0; r < DT_RANK; ++r)
      wr[r] = dtw[(dir * D_INNER + c) * DT_RANK + r];
    const float bias = dtb[dir * D_INNER + c];
    const float Dsc  = Dsv[dir * D_INNER + c];

    float An[D_STATE];
#pragma unroll
    for (int n = 0; n < D_STATE; ++n)
      An[n] = -__expf(alogs[(dir * D_INNER + c) * D_STATE + n]);

    float h[D_STATE];
#pragma unroll
    for (int n = 0; n < D_STATE; ++n) h[n] = 0.0f;

    for (int t = 0; t < LW; ++t) {
      int l;
      if      (dir == 0) l = t;
      else if (dir == 1) l = 3 * (t % 3) + t / 3;
      else if (dir == 2) l = 8 - t;
      else { const int s = 8 - t; l = 3 * (s % 3) + s / 3; }

      float draw = bias;
#pragma unroll
      for (int r = 0; r < DT_RANK; ++r) draw += xb[r * LW + l] * wr[r];
      const float delta = (draw > 20.0f) ? draw : log1pf(__expf(draw));

      const float ut = uwin[l];
      const float du = delta * ut;
      float y = 0.0f;
#pragma unroll
      for (int n = 0; n < D_STATE; ++n) {
        const float Bt = xb[(DT_RANK + n) * LW + l];
        const float Ct = xb[(DT_RANK + D_STATE + n) * LW + l];
        h[n] = h[n] * __expf(delta * An[n]) + du * Bt;
        y += h[n] * Ct;
      }
      outacc[l] += y + Dsc * ut;   // includes D*u term
    }
  }

  // layernorm stats over the 192 channels, per window position
#pragma unroll
  for (int l = 0; l < LW; ++l) red[l * D_INNER + c] = outacc[l];
  __syncthreads();
  if (c < LW) {
    float s = 0.f, s2 = 0.f;
    for (int i = 0; i < D_INNER; ++i) {
      const float v = red[c * D_INNER + i];
      s += v; s2 += v * v;
    }
    const float m  = s / D_INNER;
    const float va = s2 / D_INNER - m * m;
    mu[c] = m;
    rs[c] = rsqrtf(va + 1e-5f);
  }
  __syncthreads();

  const float w = lnw[c], b = lnb[c];
  float r = 0.0f;
#pragma unroll
  for (int l = 0; l < LW; ++l) {
    const float v = (outacc[l] - mu[l]) * rs[l] * w + b;
    r += v * convw[c * LW + l];
  }
  res[p * D_INNER + c] = r;
}

// ============================================================
// Kernel 3: out = res [2304x192] @ linear0_w^T [192x96]
// One wave32 per 16x16 tile, 6 back-to-back v_wmma_f32_16x16x32_f16.
// Both fragments are contiguous-run loads (b128).
// ============================================================
__global__ void __launch_bounds__(32)
out_gemm_wmma(const float* __restrict__ resb,  // [2304][192]
              const float* __restrict__ lw,    // [96][192]
              float* __restrict__ out)         // [2304][96]
{
  const int lane = threadIdx.x;
  const int p0 = blockIdx.x * 16;
  const int o0 = blockIdx.y * 16;
  const int ml = lane & 15;
  const int hi = lane >> 4;

  v8f acc = {};
#pragma unroll
  for (int kb = 0; kb < D_INNER; kb += 32) {
    const v16h a = load_a_frag(&resb[(p0 + ml) * D_INNER + kb], hi, 1.0f);
    const v16h b = load_b_frag(&lw[(o0 + ml) * D_INNER + kb], hi);
    acc = __builtin_amdgcn_wmma_f32_16x16x32_f16(false, a, false, b,
                                                 (short)0, acc, false, false);
  }
#pragma unroll
  for (int r = 0; r < 8; ++r)
    out[(p0 + r + 8 * hi) * OUTCH_ + o0 + ml] = acc[r];
}

// ============================================================
extern "C" void kernel_launch(void* const* d_in, const int* in_sizes, int n_in,
                              void* d_out, int out_size, void* d_ws, size_t ws_size,
                              hipStream_t stream) {
  const float* x    = (const float*)d_in[0];   // (192,48,48)
  const float* xpw  = (const float*)d_in[1];   // (4,38,192)
  const float* dtw  = (const float*)d_in[2];   // (4,192,6)
  const float* dtb  = (const float*)d_in[3];   // (4,192)
  const float* alog = (const float*)d_in[4];   // (768,16)
  const float* Dsv  = (const float*)d_in[5];   // (768)
  const float* lnw  = (const float*)d_in[6];   // (192)
  const float* lnb  = (const float*)d_in[7];   // (192)
  const float* cvw  = (const float*)d_in[8];   // (192,3,3)
  const float* lin  = (const float*)d_in[9];   // (96,192)
  float* out = (float*)d_out;                  // (1,48,48,96)

  float* xt   = (float*)d_ws;                  // 2304*192 floats
  float* xdbl = xt + HW * D_INNER;             // 152*2304 floats
  float* res  = xdbl + MROWS * HW;             // 2304*192 floats

  dim3 g0(HW / 16, D_INNER / 16);
  transpose_x<<<g0, 256, 0, stream>>>(x, xt);

  dim3 g1(HW / 16, 10);                        // 160 padded rows / 16
  xdbl_gemm_wmma<<<g1, 32, 0, stream>>>(xt, xpw, xdbl);

  scan_ln_conv<<<HW, D_INNER, 0, stream>>>(xt, xdbl, dtw, dtb, alog, Dsv,
                                           lnw, lnb, cvw, res);

  dim3 g3(HW / 16, OUTCH_ / 16);
  out_gemm_wmma<<<g3, 32, 0, stream>>>(res, lin, out);
}
